// GPRSurrogate_30013231464836
// MI455X (gfx1250) — matrix operand order repaired and checked by reading, not compiled, not run on hardware
//
#include <hip/hip_runtime.h>
#include <hip/hip_bf16.h>

typedef __attribute__((ext_vector_type(16))) __bf16 v16bf;
typedef __attribute__((ext_vector_type(8)))  float  v8f;

#define GPR_M 4096
#define GPR_N 8192
#define GPR_D 256
#define MTILE 128              // rows of X per block (8 waves x 16)
#define NSPLIT 16              // N split across blockIdx.y -> 512 blocks
#define NMACRO 64              // N rows staged in LDS per iteration
#define KSTEPS (GPR_D / 32)    // 8 WMMA k-steps of K=32
#define BUFWORDS (4 * KSTEPS * 32 * 8)   // 8192 dwords = 32 KB per buffer

union BF16Frag {
    unsigned int u[8];
    v16bf        v;
};

#if defined(__has_builtin)
#if __has_builtin(__builtin_amdgcn_cvt_pk_bf16_f32)
#define HAVE_CVT_PK_BF16 1
#endif
#endif

__device__ __forceinline__ unsigned int bf16_rne(float f) {
    unsigned int u = __float_as_uint(f);
    u += 0x7FFFu + ((u >> 16) & 1u);   // round-to-nearest-even to bf16
    return u >> 16;
}
__device__ __forceinline__ unsigned int pk_bf16(float lo, float hi) {
#ifdef HAVE_CVT_PK_BF16
    auto t = __builtin_amdgcn_cvt_pk_bf16_f32(lo, hi);  // v_cvt_pk_bf16_f32
    return __builtin_bit_cast(unsigned int, t);
#else
    return bf16_rne(lo) | (bf16_rne(hi) << 16);
#endif
}

// ---------------------------------------------------------------------------
// Kernel 1: f[m] = exp(-||X[m]||^2 / 2),  w[n] = alpha[n] * exp(-||Xt[n]||^2/2)
// ---------------------------------------------------------------------------
__global__ __launch_bounds__(256) void gpr_prep(const float* __restrict__ X,
                                                const float* __restrict__ Xt,
                                                const float* __restrict__ alpha,
                                                float* __restrict__ f,
                                                float* __restrict__ w) {
    int i = blockIdx.x * 256 + threadIdx.x;
    if (i < GPR_M) {
        const float* row = X + (size_t)i * GPR_D;
        float s = 0.f;
        #pragma unroll 4
        for (int k = 0; k < GPR_D; k += 4) {
            float4 v = *(const float4*)(row + k);
            s += v.x * v.x + v.y * v.y + v.z * v.z + v.w * v.w;
        }
        f[i] = __expf(-0.5f * s);
    }
    if (i < GPR_N) {
        const float* row = Xt + (size_t)i * GPR_D;
        float s = 0.f;
        #pragma unroll 4
        for (int k = 0; k < GPR_D; k += 4) {
            float4 v = *(const float4*)(row + k);
            s += v.x * v.x + v.y * v.y + v.z * v.z + v.w * v.w;
        }
        w[i] = alpha[i] * __expf(-0.5f * s);
    }
}

// ---------------------------------------------------------------------------
// Kernel 2: fused WMMA GEMM + exp epilogue + per-row partial reduction.
// partial[m * NSPLIT + split] = sum_{n in split} exp(X[m].Xt[n]) * w[n]
// Double-buffered LDS staging of B fragments (bf16), A fragments in VGPRs.
// ---------------------------------------------------------------------------
__global__ __launch_bounds__(256) void gpr_gemm(const float* __restrict__ X,
                                                const float* __restrict__ Xt,
                                                const float* __restrict__ w,
                                                float* __restrict__ partial) {
    // two buffers: [nsub(4)][ks(8)][lane(32)][8 dwords] each = 2 x 32 KB
    __shared__ unsigned int smem[2 * BUFWORDS];

    const int lane = threadIdx.x & 31;
    const int wave = threadIdx.x >> 5;
    const int hi   = lane >> 4;          // half-wave selector
    const int ln15 = lane & 15;
    const int m0   = blockIdx.x * MTILE;
    const int nchunk = GPR_N / NSPLIT;   // 512
    const int nstart = blockIdx.y * nchunk;
    const int niter  = nchunk / NMACRO;  // 8

    // Cooperative B staging: group g = (nsub, ks, lane) converts 16 contiguous
    // floats of X_train into 8 packed bf16 dwords in fragment layout.
    auto stage = [&](int n0, unsigned int* __restrict__ buf) {
        #pragma unroll
        for (int i = 0; i < 4; ++i) {
            int g   = threadIdx.x + 256 * i;
            int gl  = g & 31;
            int gks = (g >> 5) & 7;
            int gns = g >> 8;
            int n   = n0 + gns * 16 + (gl & 15);
            int kb  = gks * 32 + (gl >> 4) * 16;
            const float* p = Xt + (size_t)n * GPR_D + kb;
            float4 q0 = *(const float4*)(p);
            float4 q1 = *(const float4*)(p + 4);
            float4 q2 = *(const float4*)(p + 8);
            float4 q3 = *(const float4*)(p + 12);
            unsigned int* dst = buf + g * 8;
            dst[0] = pk_bf16(q0.x, q0.y);
            dst[1] = pk_bf16(q0.z, q0.w);
            dst[2] = pk_bf16(q1.x, q1.y);
            dst[3] = pk_bf16(q1.z, q1.w);
            dst[4] = pk_bf16(q2.x, q2.y);
            dst[5] = pk_bf16(q2.z, q2.w);
            dst[6] = pk_bf16(q3.x, q3.y);
            dst[7] = pk_bf16(q3.z, q3.w);
        }
    };

    // ---- Load + convert this wave's A fragments once (contiguous k order, same as B staging):
    // lanes 0-15: K = k0+{0..15}; lanes 16-31: K = k0+{16..31}.
    BF16Frag a[KSTEPS];
    {
        const int rowA = m0 + wave * 16 + ln15;
        const float* Arow = X + (size_t)rowA * GPR_D;
        #pragma unroll
        for (int ks = 0; ks < KSTEPS; ++ks) {
            const float* p0 = Arow + ks * 32 + hi * 16;
            float4 r0 = *(const float4*)(p0);
            float4 r1 = *(const float4*)(p0 + 4);
            float4 r2 = *(const float4*)(p0 + 8);
            float4 r3 = *(const float4*)(p0 + 12);
            a[ks].u[0] = pk_bf16(r0.x, r0.y);
            a[ks].u[1] = pk_bf16(r0.z, r0.w);
            a[ks].u[2] = pk_bf16(r1.x, r1.y);
            a[ks].u[3] = pk_bf16(r1.z, r1.w);
            a[ks].u[4] = pk_bf16(r2.x, r2.y);
            a[ks].u[5] = pk_bf16(r2.z, r2.w);
            a[ks].u[6] = pk_bf16(r3.x, r3.y);
            a[ks].u[7] = pk_bf16(r3.z, r3.w);
        }
    }

    float acc[8];
    #pragma unroll
    for (int r = 0; r < 8; ++r) acc[r] = 0.f;

    stage(nstart, smem);                     // prologue: fill buffer 0

    for (int it = 0; it < niter; ++it) {
        const int n0 = nstart + it * NMACRO;
        unsigned int* cur = smem + (it & 1) * BUFWORDS;
        __syncthreads();                     // staging of `cur` complete
        if (it + 1 < niter)                  // overlap: stage next tile
            stage(n0 + NMACRO, smem + ((it + 1) & 1) * BUFWORDS);

        // ---- Compute: 4 n-subtiles x 8 k-step WMMA chain each.
        #pragma unroll
        for (int s = 0; s < 4; ++s) {
            float wn = w[n0 + s * 16 + ln15];
            v8f c = {};
            #pragma unroll
            for (int ks = 0; ks < KSTEPS; ++ks) {
                BF16Frag b;
                const uint4* q =
                    (const uint4*)(cur + ((s * KSTEPS + ks) * 32 + lane) * 8);
                uint4 b0 = q[0];
                uint4 b1 = q[1];
                b.u[0] = b0.x; b.u[1] = b0.y; b.u[2] = b0.z; b.u[3] = b0.w;
                b.u[4] = b1.x; b.u[5] = b1.y; b.u[6] = b1.z; b.u[7] = b1.w;
                c = __builtin_amdgcn_wmma_f32_16x16x32_bf16(
                        false, a[ks].v, false, b.v, (short)0, c, false, false);
            }
            // Epilogue: exp on TRANS pipe (co-executes with WMMA), times w[n].
            #pragma unroll
            for (int r = 0; r < 8; ++r) acc[r] += __expf(c[r]) * wn;
        }
    }

    // ---- Reduce over the 16 lanes of each half-wave (wave32).
    // C layout: VGPR r -> lanes 0-15 hold M=r, lanes 16-31 hold M=8+r.
    #pragma unroll
    for (int r = 0; r < 8; ++r) {
        float v = acc[r];
        v += __shfl_xor(v, 1, 32);
        v += __shfl_xor(v, 2, 32);
        v += __shfl_xor(v, 4, 32);
        v += __shfl_xor(v, 8, 32);
        if (ln15 == 0) {
            int row = m0 + wave * 16 + hi * 8 + r;
            partial[(size_t)row * NSPLIT + blockIdx.y] = v;
        }
    }
}

// ---------------------------------------------------------------------------
// Kernel 3: pred[m] = f[m] * sum_splits partial[m][*]
// ---------------------------------------------------------------------------
__global__ __launch_bounds__(256) void gpr_reduce(const float* __restrict__ partial,
                                                  const float* __restrict__ f,
                                                  float* __restrict__ out) {
    int m = blockIdx.x * 256 + threadIdx.x;
    if (m < GPR_M) {
        float s = 0.f;
        #pragma unroll
        for (int i = 0; i < NSPLIT; ++i) s += partial[(size_t)m * NSPLIT + i];
        out[m] = f[m] * s;
    }
}

extern "C" void kernel_launch(void* const* d_in, const int* in_sizes, int n_in,
                              void* d_out, int out_size, void* d_ws, size_t ws_size,
                              hipStream_t stream) {
    const float* X     = (const float*)d_in[0];   // [M, D]
    const float* Xt    = (const float*)d_in[1];   // [N, D]
    const float* alpha = (const float*)d_in[2];   // [N, 1]

    float* f       = (float*)d_ws;                // M floats
    float* w       = f + GPR_M;                   // N floats
    float* partial = w + GPR_N;                   // M * NSPLIT floats

    gpr_prep<<<(GPR_N + 255) / 256, 256, 0, stream>>>(X, Xt, alpha, f, w);
    gpr_gemm<<<dim3(GPR_M / MTILE, NSPLIT), 256, 0, stream>>>(X, Xt, w, partial);
    gpr_reduce<<<(GPR_M + 255) / 256, 256, 0, stream>>>(partial, f, (float*)d_out);
}
